// MoE_31842887533081
// MI455X (gfx1250) — matrix operand order repaired and checked by reading, compiled, not run on hardware
//
#include <hip/hip_runtime.h>
#include <hip/hip_bf16.h>

// ---------------------------------------------------------------------------
// Types for CDNA5 WMMA (wave32, 16x16x32 bf16 -> f32)
// ---------------------------------------------------------------------------
typedef __attribute__((ext_vector_type(16))) __bf16 v16bf;
typedef __attribute__((ext_vector_type(8)))  __bf16 v8bf;
typedef __attribute__((ext_vector_type(8)))  float  v8f;

// gfx1250 async global->LDS copy path (ASYNCcnt), guarded so we always compile.
#if defined(__gfx1250__) && __has_builtin(__builtin_amdgcn_global_load_async_to_lds_b128) && __has_builtin(__builtin_amdgcn_s_wait_asynccnt)
#define HAVE_ASYNC_LDS 1
#define AS1 __attribute__((address_space(1)))
#define AS3 __attribute__((address_space(3)))
typedef __attribute__((__vector_size__(4 * sizeof(int)))) int gvec4i;
// issue one async 16B global->LDS copy (per lane), tracked by ASYNCcnt
// (builtin takes NON-const addrspace(1) src; C-style cast strips const)
#define ASYNC_CP16(gsrc, ldst)                                                  \
    __builtin_amdgcn_global_load_async_to_lds_b128(                             \
        (AS1 gvec4i*)(gsrc), (AS3 gvec4i*)(ldst), 0, 0)
#endif

__device__ __forceinline__ __bf16 f2bf(float f) {
    unsigned int u = __float_as_uint(f);
    unsigned int r = u + 0x7FFFu + ((u >> 16) & 1u);     // round-to-nearest-even
    unsigned short h = (unsigned short)(r >> 16);
    return __builtin_bit_cast(__bf16, h);
}

__device__ __forceinline__ v16bf cat16(v8bf lo, v8bf hi) {
    return __builtin_shufflevector(lo, hi, 0,1,2,3,4,5,6,7,8,9,10,11,12,13,14,15);
}

// Problem constants
#define TB   2
#define TS   2048
#define TD   1024
#define TE   8
#define TF   2752
#define TT   (TB*TS)          // 4096 tokens
#define TC   (TT/TE)          // 512 capacity

// GEMM tiling
#define BM   128
#define BN   64
#define BK   32
#define A_ST 40               // bf16 elems per LDS A row (80B, 16B-aligned, bank-friendly)
#define B_ST 40               // bf16 elems per LDS B(t) row

// ---------------------------------------------------------------------------
// small helpers
// ---------------------------------------------------------------------------
__global__ void zero_f32_kernel(float* p, int n) {
    int i = blockIdx.x * blockDim.x + threadIdx.x;
    if (i < n) p[i] = 0.0f;
}

__global__ void cvt_bf16_kernel(const float* __restrict__ in, __bf16* __restrict__ out, int n4) {
    int i = (blockIdx.x * blockDim.x + threadIdx.x);
    if (i >= n4) return;
    float4 v = *(const float4*)(in + i * 4);
    __bf16* o = out + i * 4;
    o[0] = f2bf(v.x); o[1] = f2bf(v.y); o[2] = f2bf(v.z); o[3] = f2bf(v.w);
}

// ---------------------------------------------------------------------------
// Gating: one wave (32 lanes) per token.  logits = x@wg, softmax, top-1;
// also the 2-way coefficient softmax for the residual combine.
// ---------------------------------------------------------------------------
__global__ __launch_bounds__(256)
void gate_kernel(const float* __restrict__ x, const float* __restrict__ wg,
                 const float* __restrict__ cw, const float* __restrict__ cb,
                 int* __restrict__ idx, float* __restrict__ gmax,
                 float* __restrict__ coef0, float* __restrict__ coef1,
                 float* __restrict__ me_acc) {
    int wave = blockIdx.x * (blockDim.x >> 5) + (threadIdx.x >> 5);
    int lane = threadIdx.x & 31;
    if (wave >= TT) return;
    const float* xr = x + (long long)wave * TD;
    float acc[TE] = {};
    float c0 = 0.f, c1 = 0.f;
    for (int d = lane; d < TD; d += 32) {
        float xv = xr[d];
        const float* w = wg + d * TE;
#pragma unroll
        for (int e = 0; e < TE; ++e) acc[e] += xv * w[e];
        c0 += xv * cw[d * 2 + 0];
        c1 += xv * cw[d * 2 + 1];
    }
#pragma unroll
    for (int off = 16; off > 0; off >>= 1) {
#pragma unroll
        for (int e = 0; e < TE; ++e) acc[e] += __shfl_xor(acc[e], off, 32);
        c0 += __shfl_xor(c0, off, 32);
        c1 += __shfl_xor(c1, off, 32);
    }
    if (lane == 0) {
        float mx = acc[0]; int am = 0;
#pragma unroll
        for (int e = 1; e < TE; ++e) if (acc[e] > mx) { mx = acc[e]; am = e; }
        float p[TE]; float s = 0.f;
#pragma unroll
        for (int e = 0; e < TE; ++e) { p[e] = __expf(acc[e] - mx); s += p[e]; }
        float inv = 1.0f / s;
#pragma unroll
        for (int e = 0; e < TE; ++e) atomicAdd(&me_acc[e], p[e] * inv);
        idx[wave]  = am;
        gmax[wave] = p[am] * inv;
        float l0 = c0 + cb[0], l1 = c1 + cb[1];
        float m2 = fmaxf(l0, l1);
        float e0 = __expf(l0 - m2), e1 = __expf(l1 - m2);
        float ci = 1.0f / (e0 + e1);
        coef0[wave] = e0 * ci;
        coef1[wave] = e1 * ci;
    }
}

// ---------------------------------------------------------------------------
// Capacity scan (token order), slot assignment + l_aux + exp_counts.
// Single workgroup; idx staged in LDS, sequential scan over 4096 ints.
// ---------------------------------------------------------------------------
__global__ __launch_bounds__(256)
void scan_kernel(const int* __restrict__ idx, const float* __restrict__ me_acc,
                 int* __restrict__ slot, int* __restrict__ tok_of,
                 int* __restrict__ kept, float* __restrict__ out_tail) {
    __shared__ int sIdx[TT];
    for (int t = threadIdx.x; t < TT; t += blockDim.x) sIdx[t] = idx[t];
    __syncthreads();
    if (threadIdx.x == 0) {
        int cntFull[TE] = {}; int cntKept[TE] = {};
        for (int t = 0; t < TT; ++t) {
            int e = sIdx[t];
            int c = cntFull[e]++;
            if (c < TC) { slot[t] = c; tok_of[e * TC + c] = t; cntKept[e] = c + 1; }
            else        { slot[t] = -1; }
        }
        float l_aux = 0.f;
        const float invT = 1.0f / (float)TT;
#pragma unroll
        for (int e = 0; e < TE; ++e) {
            kept[e] = cntKept[e];
            l_aux += (me_acc[e] * invT) * ((float)cntFull[e] * invT);
        }
        out_tail[0] = l_aux * (float)TE;
        int* oc = (int*)(out_tail + 1);
#pragma unroll
        for (int e = 0; e < TE; ++e) oc[e] = cntFull[e];
    }
}

// ---------------------------------------------------------------------------
// Gather tokens into expert buffers [E*C, D], f32 -> bf16, zero unused slots.
// ---------------------------------------------------------------------------
__global__ __launch_bounds__(128)
void dispatch_kernel(const float* __restrict__ x, const int* __restrict__ tok_of,
                     const int* __restrict__ kept, __bf16* __restrict__ xe) {
    int row = blockIdx.x;                   // e*C + c
    int e = row / TC, c = row % TC;
    bool valid = c < kept[e];
    int t = valid ? tok_of[row] : 0;
    const float* src = x + (long long)t * TD;
    __bf16* dst = xe + (long long)row * TD;
    for (int d = threadIdx.x; d < TD; d += blockDim.x)
        dst[d] = valid ? f2bf(src[d]) : f2bf(0.0f);
}

// ---------------------------------------------------------------------------
// Stage 1: H = silu(X@W1) * (X@W3),   X bf16 [M,K] row-major, W f32 [K,N],
// H bf16 [M,N].  8 waves/WG, WG tile 128x64, wave tile 64x16 per matrix.
// A tile copied global->LDS via GLOBAL_LOAD_ASYNC_TO_LDS (overlaps the
// f32->bf16 weight conversion); f32 weights read from HBM exactly once.
// ---------------------------------------------------------------------------
__global__ __launch_bounds__(256)
void gemm_swiglu_kernel(const __bf16* __restrict__ X,
                        const float* __restrict__ W1, const float* __restrict__ W3,
                        __bf16* __restrict__ H, int M, int K, int N,
                        long long xBatch, long long wBatch, long long hBatch) {
    __shared__ __bf16 As[BM * A_ST];
    __shared__ __bf16 Bs1[BN * B_ST];
    __shared__ __bf16 Bs3[BN * B_ST];

    const int b = blockIdx.z;
    X  += (long long)b * xBatch;
    W1 += (long long)b * wBatch;
    W3 += (long long)b * wBatch;
    H  += (long long)b * hBatch;

    const int n0 = blockIdx.x * BN;
    const int m0 = blockIdx.y * BM;
    const int tid = threadIdx.x;
    const int wave = tid >> 5, lane = tid & 31;
    const int wm = wave & 1;        // M half (64 rows)
    const int wn = wave >> 1;       // N quarter (16 cols)
    const int lm = lane & 15, lh = lane >> 4;

    v8f acc1[4] = {}; v8f acc3[4] = {};

    for (int k0 = 0; k0 < K; k0 += BK) {
        // ---- stage X tile 128x32 bf16 into LDS (32B per thread) ----
        {
            int row = tid >> 1;
            int ch  = (tid & 1) * 16;
            const __bf16* src = X + (long long)(m0 + row) * K + k0 + ch;
#ifdef HAVE_ASYNC_LDS
            ASYNC_CP16(src,     &As[row * A_ST + ch]);
            ASYNC_CP16(src + 8, &As[row * A_ST + ch + 8]);
#else
            uint4 d0 = *(const uint4*)(src);
            uint4 d1 = *(const uint4*)(src + 8);
            *(uint4*)(&As[row * A_ST + ch])     = d0;
            *(uint4*)(&As[row * A_ST + ch + 8]) = d1;
#endif
        }
        // ---- stage W1/W3 tiles 32x64 f32 -> bf16 transposed [n][k] ----
        {
            int kr = tid >> 3;            // 0..31
            int nc = (tid & 7) * 8;       // 0..56
            const float* s1 = W1 + (long long)(k0 + kr) * N + n0 + nc;
            const float* s3 = W3 + (long long)(k0 + kr) * N + n0 + nc;
            __builtin_prefetch(s1 + (long long)BK * N, 0, 1);   // global_prefetch_b8
            __builtin_prefetch(s3 + (long long)BK * N, 0, 1);
            float4 a = *(const float4*)(s1);
            float4 bb = *(const float4*)(s1 + 4);
            float4 c = *(const float4*)(s3);
            float4 d = *(const float4*)(s3 + 4);
            float v1[8] = {a.x,a.y,a.z,a.w,bb.x,bb.y,bb.z,bb.w};
            float v3[8] = {c.x,c.y,c.z,c.w,d.x,d.y,d.z,d.w};
#pragma unroll
            for (int j = 0; j < 8; ++j) {
                Bs1[(nc + j) * B_ST + kr] = f2bf(v1[j]);
                Bs3[(nc + j) * B_ST + kr] = f2bf(v3[j]);
            }
        }
#ifdef HAVE_ASYNC_LDS
        __builtin_amdgcn_s_wait_asynccnt(0);
#endif
        __syncthreads();

        // ---- B fragments: lane lm = column, lh selects K 8h..8h+7 and 16+8h..16+8h+7 ----
        const __bf16* bp1 = &Bs1[(wn * 16 + lm) * B_ST + lh * 8];
        const __bf16* bp3 = &Bs3[(wn * 16 + lm) * B_ST + lh * 8];
        v16bf fb1 = cat16(*(const v8bf*)bp1, *(const v8bf*)(bp1 + 16));
        v16bf fb3 = cat16(*(const v8bf*)bp3, *(const v8bf*)(bp3 + 16));

#pragma unroll
        for (int mt = 0; mt < 4; ++mt) {
            const __bf16* ap = &As[(wm * 64 + mt * 16 + lm) * A_ST + 8 * lh];
            v16bf fa = cat16(*(const v8bf*)ap, *(const v8bf*)(ap + 16));
            acc1[mt] = __builtin_amdgcn_wmma_f32_16x16x32_bf16(
                false, fa, false, fb1, (short)0, acc1[mt], false, false);
            acc3[mt] = __builtin_amdgcn_wmma_f32_16x16x32_bf16(
                false, fa, false, fb3, (short)0, acc3[mt], false, false);
        }
        __syncthreads();
    }

    // ---- epilogue: silu(a)*c -> bf16; lanes 0..15 contiguous in n ----
    const int ncol = n0 + wn * 16 + lm;
#pragma unroll
    for (int mt = 0; mt < 4; ++mt) {
#pragma unroll
        for (int r = 0; r < 8; ++r) {
            int mrow = m0 + wm * 64 + mt * 16 + r + 8 * lh;
            float a = acc1[mt][r];
            float hv = (a / (1.0f + __expf(-a))) * acc3[mt][r];
            H[(long long)mrow * N + ncol] = f2bf(hv);
        }
    }
}

// ---------------------------------------------------------------------------
// Stage 2: Y = H @ W2,  H bf16 [M,K], W2 f32 [K,N], Y f32 [M,N].
// ---------------------------------------------------------------------------
__global__ __launch_bounds__(256)
void gemm_out_kernel(const __bf16* __restrict__ Hin, const float* __restrict__ W2,
                     float* __restrict__ Y, int M, int K, int N,
                     long long hBatch, long long wBatch, long long yBatch) {
    __shared__ __bf16 As[BM * A_ST];
    __shared__ __bf16 Bs[BN * B_ST];

    const int b = blockIdx.z;
    Hin += (long long)b * hBatch;
    W2  += (long long)b * wBatch;
    Y   += (long long)b * yBatch;

    const int n0 = blockIdx.x * BN;
    const int m0 = blockIdx.y * BM;
    const int tid = threadIdx.x;
    const int wave = tid >> 5, lane = tid & 31;
    const int wm = wave & 1;
    const int wn = wave >> 1;
    const int lm = lane & 15, lh = lane >> 4;

    v8f acc[4] = {};

    for (int k0 = 0; k0 < K; k0 += BK) {
        {
            int row = tid >> 1;
            int ch  = (tid & 1) * 16;
            const __bf16* src = Hin + (long long)(m0 + row) * K + k0 + ch;
#ifdef HAVE_ASYNC_LDS
            ASYNC_CP16(src,     &As[row * A_ST + ch]);
            ASYNC_CP16(src + 8, &As[row * A_ST + ch + 8]);
#else
            uint4 d0 = *(const uint4*)(src);
            uint4 d1 = *(const uint4*)(src + 8);
            *(uint4*)(&As[row * A_ST + ch])     = d0;
            *(uint4*)(&As[row * A_ST + ch + 8]) = d1;
#endif
        }
        {
            int kr = tid >> 3;
            int nc = (tid & 7) * 8;
            const float* s2 = W2 + (long long)(k0 + kr) * N + n0 + nc;
            __builtin_prefetch(s2 + (long long)BK * N, 0, 1);
            float4 a = *(const float4*)(s2);
            float4 bb = *(const float4*)(s2 + 4);
            float v2[8] = {a.x,a.y,a.z,a.w,bb.x,bb.y,bb.z,bb.w};
#pragma unroll
            for (int j = 0; j < 8; ++j)
                Bs[(nc + j) * B_ST + kr] = f2bf(v2[j]);
        }
#ifdef HAVE_ASYNC_LDS
        __builtin_amdgcn_s_wait_asynccnt(0);
#endif
        __syncthreads();

        const __bf16* bp = &Bs[(wn * 16 + lm) * B_ST + lh * 8];
        v16bf fb = cat16(*(const v8bf*)bp, *(const v8bf*)(bp + 16));

#pragma unroll
        for (int mt = 0; mt < 4; ++mt) {
            const __bf16* ap = &As[(wm * 64 + mt * 16 + lm) * A_ST + 8 * lh];
            v16bf fa = cat16(*(const v8bf*)ap, *(const v8bf*)(ap + 16));
            acc[mt] = __builtin_amdgcn_wmma_f32_16x16x32_bf16(
                false, fa, false, fb, (short)0, acc[mt], false, false);
        }
        __syncthreads();
    }

    const int ncol = n0 + wn * 16 + lm;
#pragma unroll
    for (int mt = 0; mt < 4; ++mt) {
#pragma unroll
        for (int r = 0; r < 8; ++r) {
            int mrow = m0 + wm * 64 + mt * 16 + r + 8 * lh;
            Y[(long long)mrow * N + ncol] = acc[mt][r];
        }
    }
}

// ---------------------------------------------------------------------------
// Final combine: out = gate*Ye (undropped) * coef0 + mlp * coef1
// ---------------------------------------------------------------------------
__global__ __launch_bounds__(128)
void combine_kernel(const float* __restrict__ Ye, const float* __restrict__ Mlp,
                    const int* __restrict__ idx, const int* __restrict__ slot,
                    const float* __restrict__ gmax, const float* __restrict__ coef0,
                    const float* __restrict__ coef1, float* __restrict__ out) {
    int t = blockIdx.x;
    int s = slot[t];
    float g  = (s >= 0) ? gmax[t] : 0.0f;
    long long yoff = (s >= 0) ? (long long)(idx[t] * TC + s) * TD : 0;
    float c0 = coef0[t], c1 = coef1[t];
    const float* mSrc = Mlp + (long long)t * TD;
    float* o = out + (long long)t * TD;
    for (int d = threadIdx.x; d < TD; d += blockDim.x) {
        float moe = (s >= 0) ? g * Ye[yoff + d] : 0.0f;
        o[d] = moe * c0 + mSrc[d] * c1;
    }
}

// ---------------------------------------------------------------------------
// Host launcher
// ---------------------------------------------------------------------------
extern "C" void kernel_launch(void* const* d_in, const int* in_sizes, int n_in,
                              void* d_out, int out_size, void* d_ws, size_t ws_size,
                              hipStream_t stream) {
    const float* x   = (const float*)d_in[0];
    const float* wg  = (const float*)d_in[1];
    const float* w1  = (const float*)d_in[2];
    const float* w3  = (const float*)d_in[3];
    const float* w2  = (const float*)d_in[4];
    const float* rw1 = (const float*)d_in[5];
    const float* rw3 = (const float*)d_in[6];
    const float* rw2 = (const float*)d_in[7];
    const float* cw  = (const float*)d_in[8];
    const float* cb  = (const float*)d_in[9];
    float* out = (float*)d_out;

    // workspace carve-out (256B aligned blocks)
    char* p = (char*)d_ws;
    auto carve = [&](size_t bytes) { char* r = p; p += (bytes + 255) & ~(size_t)255; return r; };
    __bf16* xall  = (__bf16*)carve(sizeof(__bf16) * (size_t)TT * TD);        // 8 MB
    __bf16* xe    = (__bf16*)carve(sizeof(__bf16) * (size_t)TE * TC * TD);   // 8 MB
    __bf16* Hbuf  = (__bf16*)carve(sizeof(__bf16) * (size_t)TT * TF);        // 22.5 MB (shared expert/residual)
    float*  Ye    = (float*) carve(sizeof(float)  * (size_t)TE * TC * TD);   // 16.8 MB
    float*  Mlp   = (float*) carve(sizeof(float)  * (size_t)TT * TD);        // 16.8 MB
    int*    idx   = (int*)   carve(sizeof(int) * TT);
    int*    slot  = (int*)   carve(sizeof(int) * TT);
    float*  gmax  = (float*) carve(sizeof(float) * TT);
    float*  coef0 = (float*) carve(sizeof(float) * TT);
    float*  coef1 = (float*) carve(sizeof(float) * TT);
    int*    tokof = (int*)   carve(sizeof(int) * TE * TC);
    int*    kept  = (int*)   carve(sizeof(int) * TE);
    float*  meacc = (float*) carve(sizeof(float) * TE);

    // 0) zero the me accumulator (ws not re-poisoned between replays)
    zero_f32_kernel<<<1, 64, 0, stream>>>(meacc, TE);

    // 1) hidden f32 -> bf16 for the residual-MLP GEMM
    cvt_bf16_kernel<<<((size_t)TT * TD / 4 + 255) / 256, 256, 0, stream>>>(x, xall, TT * TD / 4);

    // 2) gating + coefficient softmax (one wave per token)
    gate_kernel<<<TT / 8, 256, 0, stream>>>(x, wg, cw, cb, idx, gmax, coef0, coef1, meacc);

    // 3) capacity scan; writes l_aux + exp_counts into output tail
    scan_kernel<<<1, 256, 0, stream>>>(idx, meacc, slot, tokof, kept, out + (size_t)TT * TD);

    // 4) dispatch/gather to [E*C, D] bf16
    dispatch_kernel<<<TE * TC, 128, 0, stream>>>(x, tokof, kept, xe);

    // 5) expert SwiGLU stage 1:  [C,D] x [D,F] per expert
    dim3 g1e(TF / BN, TC / BM, TE);
    gemm_swiglu_kernel<<<g1e, 256, 0, stream>>>(xe, w1, w3, Hbuf, TC, TD, TF,
                                                (long long)TC * TD, (long long)TD * TF,
                                                (long long)TC * TF);
    // 6) expert stage 2: [C,F] x [F,D] per expert
    dim3 g2e(TD / BN, TC / BM, TE);
    gemm_out_kernel<<<g2e, 256, 0, stream>>>(Hbuf, w2, Ye, TC, TF, TD,
                                             (long long)TC * TF, (long long)TF * TD,
                                             (long long)TC * TD);
    // 7) residual MLP stage 1 over all tokens (reuses Hbuf, stream-ordered)
    dim3 g1r(TF / BN, TT / BM, 1);
    gemm_swiglu_kernel<<<g1r, 256, 0, stream>>>(xall, rw1, rw3, Hbuf, TT, TD, TF, 0, 0, 0);
    // 8) residual MLP stage 2
    dim3 g2r(TD / BN, TT / BM, 1);
    gemm_out_kernel<<<g2r, 256, 0, stream>>>(Hbuf, rw2, Mlp, TT, TF, TD, 0, 0, 0);

    // 9) weighted combine
    combine_kernel<<<TT, 128, 0, stream>>>(Ye, Mlp, idx, slot, gmax, coef0, coef1, out);
}
